// SharedMoEBlock_82411832475880
// MI455X (gfx1250) — compile-verified, matrix operand order checked by reading, not run on hardware
//
#include <hip/hip_runtime.h>

// ---------------- problem constants ----------------
constexpr int Tn = 4096;   // tokens (B*S)
constexpr int Dn = 1024;   // model dim
constexpr int Hn = 2048;   // hidden dim
constexpr int On = 1024;   // output dim
constexpr int En = 8;      // experts
constexpr int CAP = Tn;    // per-expert list capacity

constexpr int LSTR = Hn + 8;                 // LDS act row stride (bf16 elems), +16B avoids bank conflicts
constexpr int SMEM_BYTES = 32 * LSTR * 2 + 32 * 4 + 32 * 4; // act + tok + weight

typedef __bf16 v16bf __attribute__((ext_vector_type(16)));
typedef __bf16 v8bf  __attribute__((ext_vector_type(8)));
typedef float  v8f   __attribute__((ext_vector_type(8)));
typedef __bf16 v4bf  __attribute__((ext_vector_type(4)));

// fp32 -> bf16 (round-to-nearest-even), bit-level so we never need scalar __bf16 math
__device__ __forceinline__ unsigned short f2bf(float f) {
  unsigned int u = __float_as_uint(f);
  u += 0x7fffu + ((u >> 16) & 1u);
  return (unsigned short)(u >> 16);
}

__device__ __forceinline__ v8f zero8() {
  v8f c; 
#pragma unroll
  for (int i = 0; i < 8; ++i) c[i] = 0.f;
  return c;
}

// Load a 16-element bf16 WMMA A/B fragment: 8 elems at p, 8 elems at p+16
// (per-lane K layout of V_WMMA_*_16X16X32: koff handled by caller)
__device__ __forceinline__ v16bf load_ab(const unsigned short* p) {
  v8bf lo = *reinterpret_cast<const v8bf*>(p);
  v8bf hi = *reinterpret_cast<const v8bf*>(p + 16);
  return __builtin_shufflevector(lo, hi, 0,1,2,3,4,5,6,7,8,9,10,11,12,13,14,15);
}

__device__ __forceinline__ v8f wmma_bf16(v16bf a, v16bf b, v8f c) {
  // (neg_a, A, neg_b, B, c_mod, C, reuse_a, reuse_b)
  return __builtin_amdgcn_wmma_f32_16x16x32_bf16(false, a, false, b, (short)0, c, false, false);
}

// ---------------- fp32 -> bf16 bulk convert ----------------
__global__ void cvt_kernel(const float* __restrict__ s, unsigned short* __restrict__ d, int n) {
  int i = (blockIdx.x * blockDim.x + threadIdx.x) * 4;
  if (i >= n) return;
  float4 v = *reinterpret_cast<const float4*>(s + i);
  unsigned short o0 = f2bf(v.x), o1 = f2bf(v.y), o2 = f2bf(v.z), o3 = f2bf(v.w);
  unsigned long long pk = (unsigned long long)o0 | ((unsigned long long)o1 << 16) |
                          ((unsigned long long)o2 << 32) | ((unsigned long long)o3 << 48);
  *reinterpret_cast<unsigned long long*>(d + i) = pk;
}

// ---------------- router: logits, softmax, top-2, compacted per-expert lists ----------------
__global__ void router_kernel(const float* __restrict__ X, const float* __restrict__ RW,
                              int* __restrict__ cnt, int* __restrict__ list,
                              float* __restrict__ wl) {
  int gid  = blockIdx.x * blockDim.x + threadIdx.x;
  int t    = gid >> 5;        // wave per token (wave32)
  int lane = threadIdx.x & 31;
  if (t >= Tn) return;

  float acc[En];
#pragma unroll
  for (int e = 0; e < En; ++e) acc[e] = 0.f;

  const float* xr = X + (size_t)t * Dn;
  for (int d = lane; d < Dn; d += 32) {
    float xv = xr[d];
#pragma unroll
    for (int e = 0; e < En; ++e) acc[e] += xv * RW[e * Dn + d];
  }
#pragma unroll
  for (int e = 0; e < En; ++e) {
#pragma unroll
    for (int off = 16; off > 0; off >>= 1) acc[e] += __shfl_xor(acc[e], off, 32);
  }
  if (lane == 0) {
    float m = acc[0];
#pragma unroll
    for (int e = 1; e < En; ++e) m = fmaxf(m, acc[e]);
    float p[En];
#pragma unroll
    for (int e = 0; e < En; ++e) p[e] = expf(acc[e] - m);
    // top-2 of p[] (softmax denominator cancels in the renormalization)
    int i1 = 0;
#pragma unroll
    for (int e = 1; e < En; ++e) if (p[e] > p[i1]) i1 = e;
    int i2 = (i1 == 0) ? 1 : 0;
#pragma unroll
    for (int e = 0; e < En; ++e) if (e != i1 && p[e] > p[i2]) i2 = e;
    float wsum = p[i1] + p[i2];
    float w1 = p[i1] / wsum, w2 = p[i2] / wsum;
    int p1 = atomicAdd(&cnt[i1], 1);
    list[i1 * CAP + p1] = t; wl[i1 * CAP + p1] = w1;
    int p2 = atomicAdd(&cnt[i2], 1);
    list[i2 * CAP + p2] = t; wl[i2 * CAP + p2] = w2;
  }
}

// ---------------- fused SwiGLU block: (X @ Wg^T, X @ Wu^T) -> silu*up (LDS) -> @ Wd^T ----------------
// One workgroup = 32 token rows. 8 waves; per wave per iteration a 32Mx64N WMMA block.
// ROUTED=false: rows are tile*32.., plain stores (initializes Out).
// ROUTED=true : rows gathered from per-expert list, weighted atomicAdd into Out.
template <bool ROUTED>
__global__ __launch_bounds__(256) void swiglu_block(
    const unsigned short* __restrict__ X,    // [Tn,Dn]  bf16
    const unsigned short* __restrict__ Wg,   // [(E,)Hn,Dn] bf16
    const unsigned short* __restrict__ Wu,   // [(E,)Hn,Dn] bf16
    const unsigned short* __restrict__ Wd,   // [(E,)On,Hn] bf16
    const int* __restrict__ cnt, const int* __restrict__ list,
    const float* __restrict__ wl, float* __restrict__ Out) {
  extern __shared__ char smem_raw[];
  unsigned short* sAct = (unsigned short*)smem_raw;                 // [32][LSTR]
  int*   sTok = (int*)(smem_raw + 32 * LSTR * 2);                   // [32]
  float* sW   = (float*)(sTok + 32);                                // [32]

  int tid  = threadIdx.x;
  int wave = tid >> 5;
  int lane = tid & 31;
  int r    = lane & 15;
  int koff = (lane >> 4) << 3;   // 0 or 8 : K sub-offset per ISA A/B layout
  int hi   = (lane >> 4) << 3;   // 0 or 8 : M sub-offset for C/D layout
  int nl   = lane & 15;

  int e = 0, tile = blockIdx.x, count = Tn;
  const unsigned short *wg = Wg, *wu = Wu, *wd = Wd;
  if (ROUTED) {
    e    = blockIdx.x / (Tn / 32);
    tile = blockIdx.x % (Tn / 32);
    count = cnt[e];
    if (tile * 32 >= count) return;           // uniform across block
    wg = Wg + (size_t)e * Hn * Dn;
    wu = Wu + (size_t)e * Hn * Dn;
    wd = Wd + (size_t)e * On * Hn;
  }
  int base = tile * 32;

  if (tid < 32) {
    if (ROUTED) {
      int idx = base + tid;
      int ci  = (idx < count) ? idx : (count - 1);
      sTok[tid] = list[e * CAP + ci];
      sW[tid]   = (idx < count) ? wl[e * CAP + ci] : 0.f;
    } else {
      sTok[tid] = base + tid;
      sW[tid]   = 1.f;
    }
  }
  __syncthreads();

  // -------- phase 1: gate & up GEMM + silu*up -> LDS --------
  const unsigned short* aP0 = X + (size_t)sTok[r]      * Dn;
  const unsigned short* aP1 = X + (size_t)sTok[16 + r] * Dn;

  for (int it = 0; it < Hn / (8 * 64); ++it) {          // 4 iterations
    int nb = (it * 8 + wave) * 64;
    const unsigned short* bgP[4];
    const unsigned short* buP[4];
#pragma unroll
    for (int nt = 0; nt < 4; ++nt) {
      int n = nb + nt * 16 + r;
      bgP[nt] = wg + (size_t)n * Dn;
      buP[nt] = wu + (size_t)n * Dn;
    }
    v8f cg[2][4], cu[2][4];
#pragma unroll
    for (int m = 0; m < 2; ++m)
#pragma unroll
      for (int nt = 0; nt < 4; ++nt) { cg[m][nt] = zero8(); cu[m][nt] = zero8(); }

#pragma unroll 4
    for (int kb = 0; kb < Dn / 32; ++kb) {              // 32 K-steps
      int o = kb * 32 + koff;
      v16bf a0 = load_ab(aP0 + o);
      v16bf a1 = load_ab(aP1 + o);
#pragma unroll
      for (int nt = 0; nt < 4; ++nt) {
        v16bf bg = load_ab(bgP[nt] + o);
        cg[0][nt] = wmma_bf16(a0, bg, cg[0][nt]);
        cg[1][nt] = wmma_bf16(a1, bg, cg[1][nt]);
        v16bf bu = load_ab(buP[nt] + o);
        cu[0][nt] = wmma_bf16(a0, bu, cu[0][nt]);
        cu[1][nt] = wmma_bf16(a1, bu, cu[1][nt]);
      }
    }
    // epilogue: silu(g)*u -> bf16 -> LDS
#pragma unroll
    for (int m = 0; m < 2; ++m)
#pragma unroll
      for (int nt = 0; nt < 4; ++nt)
#pragma unroll
        for (int g = 0; g < 8; ++g) {
          int ml = m * 16 + hi + g;
          int nn = nb + nt * 16 + nl;
          float gv = cg[m][nt][g];
          float uv = cu[m][nt][g];
          float s  = (gv / (1.f + expf(-gv))) * uv;
          sAct[ml * LSTR + nn] = f2bf(s);
        }
  }
  __syncthreads();

  // -------- phase 2: down GEMM (A from LDS) --------
  const unsigned short* aL0 = sAct + (size_t)r        * LSTR;
  const unsigned short* aL1 = sAct + (size_t)(16 + r) * LSTR;

  for (int it = 0; it < On / (8 * 64); ++it) {          // 2 iterations
    int nb = (it * 8 + wave) * 64;
    const unsigned short* bdP[4];
#pragma unroll
    for (int nt = 0; nt < 4; ++nt) {
      int n = nb + nt * 16 + r;
      bdP[nt] = wd + (size_t)n * Hn;
    }
    v8f cd[2][4];
#pragma unroll
    for (int m = 0; m < 2; ++m)
#pragma unroll
      for (int nt = 0; nt < 4; ++nt) cd[m][nt] = zero8();

#pragma unroll 4
    for (int kb = 0; kb < Hn / 32; ++kb) {              // 64 K-steps
      int o = kb * 32 + koff;
      v16bf a0 = load_ab(aL0 + o);
      v16bf a1 = load_ab(aL1 + o);
#pragma unroll
      for (int nt = 0; nt < 4; ++nt) {
        v16bf bd = load_ab(bdP[nt] + o);
        cd[0][nt] = wmma_bf16(a0, bd, cd[0][nt]);
        cd[1][nt] = wmma_bf16(a1, bd, cd[1][nt]);
      }
    }
#pragma unroll
    for (int m = 0; m < 2; ++m)
#pragma unroll
      for (int nt = 0; nt < 4; ++nt)
#pragma unroll
        for (int g = 0; g < 8; ++g) {
          int ml  = m * 16 + hi + g;
          int tok = sTok[ml];
          int nn  = nb + nt * 16 + nl;
          float v = cd[m][nt][g];
          if (ROUTED) {
            atomicAdd(Out + (size_t)tok * On + nn, sW[ml] * v);
          } else {
            Out[(size_t)tok * On + nn] = v;
          }
        }
  }
}

// ---------------- host launch ----------------
extern "C" void kernel_launch(void* const* d_in, const int* in_sizes, int n_in,
                              void* d_out, int out_size, void* d_ws, size_t ws_size,
                              hipStream_t stream) {
  const float* hx = (const float*)d_in[0];  // [B,S,D]
  const float* rw = (const float*)d_in[1];  // [E,D]
  const float* sg = (const float*)d_in[2];  // [H,D]
  const float* su = (const float*)d_in[3];  // [H,D]
  const float* sd = (const float*)d_in[4];  // [O,H]
  const float* eg = (const float*)d_in[5];  // [E,H,D]
  const float* eu = (const float*)d_in[6];  // [E,H,D]
  const float* ed = (const float*)d_in[7];  // [E,O,H]
  float* out = (float*)d_out;

  // workspace layout (bf16 arrays then router state)
  unsigned short* p = (unsigned short*)d_ws;
  unsigned short* Xb  = p; p += (size_t)Tn * Dn;
  unsigned short* sgb = p; p += (size_t)Hn * Dn;
  unsigned short* sub = p; p += (size_t)Hn * Dn;
  unsigned short* sdb = p; p += (size_t)On * Hn;
  unsigned short* egb = p; p += (size_t)En * Hn * Dn;
  unsigned short* eub = p; p += (size_t)En * Hn * Dn;
  unsigned short* edb = p; p += (size_t)En * On * Hn;
  int*   cnt  = (int*)p;
  int*   list = cnt + En;
  float* wlp  = (float*)(list + (size_t)En * CAP);

  hipMemsetAsync(cnt, 0, En * sizeof(int), stream);

  auto cvt = [&](const float* s, unsigned short* d, size_t n) {
    int blocks = (int)((n / 4 + 255) / 256);
    cvt_kernel<<<blocks, 256, 0, stream>>>(s, d, (int)n);
  };
  cvt(hx, Xb,  (size_t)Tn * Dn);
  cvt(sg, sgb, (size_t)Hn * Dn);
  cvt(su, sub, (size_t)Hn * Dn);
  cvt(sd, sdb, (size_t)On * Hn);
  cvt(eg, egb, (size_t)En * Hn * Dn);
  cvt(eu, eub, (size_t)En * Hn * Dn);
  cvt(ed, edb, (size_t)En * On * Hn);

  router_kernel<<<(Tn * 32) / 256, 256, 0, stream>>>(hx, rw, cnt, list, wlp);

  // shared expert: writes (initializes) all of Out
  swiglu_block<false><<<Tn / 32, 256, SMEM_BYTES, stream>>>(
      Xb, sgb, sub, sdb, nullptr, nullptr, nullptr, out);

  // routed experts: weighted atomic accumulate into Out
  swiglu_block<true><<<En * (Tn / 32), 256, SMEM_BYTES, stream>>>(
      Xb, egb, eub, edb, cnt, list, wlp, out);
}
